// _Interactions_23021024707092
// MI455X (gfx1250) — matrix operand order repaired and checked by reading, not compiled, not run on hardware
//
#include <hip/hip_runtime.h>
#include <math.h>

// ---------------------------------------------------------------------------
// MPNN interaction block for MI455X (gfx1250, wave32, WMMA).
//
// msg[e,g] = sum_f x_j[e,f] * w_e[e,f,g],  w_e = reshape(t[e]@nn2_w + nn2_b)
//          = sum_k t[e,k] * (x_j[e] @ W2_k)[g]  +  (x_j[e] @ B2)[g]
// so the per-edge 64x64 matvec becomes a dense GEMM against a fixed
// [64 x 1024] f16 matrix (128 KB, L2-resident) + an in-register k-contract.
// All GEMMs use v_wmma_f32_16x16x32_f16 (f16 in, f32 accumulate).
// ---------------------------------------------------------------------------

typedef __attribute__((ext_vector_type(16))) _Float16 v16h;
typedef __attribute__((ext_vector_type(8)))  _Float16 v8h;
typedef __attribute__((ext_vector_type(8)))  float    v8f;

#define NF 64
#define ETILES 2   // edge tiles (of 16 edges) per wave: reuses B fragments

__device__ __forceinline__ v8f wmma_f16(v16h a, v16h b, v8f c) {
  // 8 args: (neg_a, A, neg_b, B, c_mod, C, reuse_a, reuse_b)
  return __builtin_amdgcn_wmma_f32_16x16x32_f16(false, a, false, b, (short)0, c,
                                                false, false);
}

// A fragment (16x32 f16), rows contiguous starting at r0, K window k0..k0+31.
// ISA layout: lanes 0-15 -> M=lane, K in {kb..kb+7, kb+16..kb+23}, kb = 0;
//             lanes 16-31 -> kb = 8.
__device__ __forceinline__ v16h frag_a_rows(const _Float16* base, int ld,
                                            int r0, int k0, int lane) {
  int m  = lane & 15;
  int kb = k0 + ((lane & 16) ? 8 : 0);
  const _Float16* p = base + (size_t)(r0 + m) * ld + kb;
  v8h lo = *(const v8h*)(p);       // K = kb .. kb+7
  v8h hi = *(const v8h*)(p + 16);  // K = kb+16 .. kb+23
  v16h r;
#pragma unroll
  for (int i = 0; i < 8; ++i) { r[i] = lo[i]; r[i + 8] = hi[i]; }
  return r;
}

// A fragment with gathered rows (row indices in LDS).
__device__ __forceinline__ v16h frag_a_gather(const _Float16* base,
                                              const int* rows, int k0, int lane) {
  int m  = lane & 15;
  int kb = k0 + ((lane & 16) ? 8 : 0);
  const _Float16* p = base + (size_t)rows[m] * NF + kb;
  v8h lo = *(const v8h*)(p);
  v8h hi = *(const v8h*)(p + 16);
  v16h r;
#pragma unroll
  for (int i = 0; i < 8; ++i) { r[i] = lo[i]; r[i + 8] = hi[i]; }
  return r;
}

// B fragment (32x16 f16) from column-major-packed weights Wcm[col*64 + k].
// Lanes 0-15 -> N=lane, K in {kb..kb+7, kb+16..kb+23}, kb = k0; lanes 16-31 -> kb = k0+8.
__device__ __forceinline__ v16h frag_b_cm(const _Float16* base, int c0, int k0,
                                          int lane) {
  int n  = lane & 15;
  int kb = k0 + ((lane & 16) ? 8 : 0);
  const _Float16* p = base + (size_t)(c0 + n) * NF + kb;
  v8h lo = *(const v8h*)(p);
  v8h hi = *(const v8h*)(p + 16);
  v16h r;
#pragma unroll
  for (int i = 0; i < 8; ++i) { r[i] = lo[i]; r[i + 8] = hi[i]; }
  return r;
}

// ------------------------------- prep kernels ------------------------------

__global__ void prep_weights(const float* lin0_w, const float* root_w,
                             const float* nn2_w, const float* nn2_b,
                             const float* w_ih, const float* w_hh,
                             _Float16* lin0_cm, _Float16* root_cm,
                             _Float16* W2p_cm, _Float16* B2_cm,
                             _Float16* wih_cm, _Float16* whh_cm) {
  int tid = blockIdx.x * blockDim.x + threadIdx.x;
  int nth = gridDim.x * blockDim.x;
  // 64x64 weights, column-major f16: Wcm[o*64+i] = W[i*64+o]
  for (int idx = tid; idx < 64 * 64; idx += nth) {
    int o = idx >> 6, i = idx & 63;
    lin0_cm[idx] = (_Float16)lin0_w[i * 64 + o];
    root_cm[idx] = (_Float16)root_w[i * 64 + o];
    B2_cm[idx]   = (_Float16)nn2_b[i * 64 + o];  // B2[f,g] = nn2_b[f*64+g]
  }
  // GRU weights [64,192] -> cm [192][64]
  for (int idx = tid; idx < 192 * 64; idx += nth) {
    int o = idx >> 6, i = idx & 63;
    wih_cm[idx] = (_Float16)w_ih[i * 192 + o];
    whh_cm[idx] = (_Float16)w_hh[i * 192 + o];
  }
  // W2p_cm[(k*64+g)*64 + f] = nn2_w[k*4096 + f*64 + g]
  for (int idx = tid; idx < 1024 * 64; idx += nth) {
    int c = idx >> 6, f = idx & 63;
    int k = c >> 6, g = c & 63;
    W2p_cm[idx] = (_Float16)nn2_w[k * 4096 + f * 64 + g];
  }
}

__global__ void cvt_f16(const float* in, _Float16* out, int n) {
  int i = blockIdx.x * blockDim.x + threadIdx.x;
  if (i < n) out[i] = (_Float16)in[i];
}

__global__ void zero_f32(float* p, int n) {
  int i = blockIdx.x * blockDim.x + threadIdx.x;
  if (i < n) p[i] = 0.0f;
}

__global__ void count_dst(const int* dst, float* cnt, int nE) {
  int e = blockIdx.x * blockDim.x + threadIdx.x;
  if (e < nE) atomicAdd(&cnt[dst[e]], 1.0f);
}

// t = relu( relu(edge_attr @ short_w + short_b) @ nn1_w + nn1_b )   [E,16]
__global__ void edge_feat(const float* edge_attr, const float* short_w,
                          const float* short_b, const float* nn1_w,
                          const float* nn1_b, float* tfeat, int nE) {
  int e = blockIdx.x * blockDim.x + threadIdx.x;
  if (e >= nE) return;
  float ga[5];
#pragma unroll
  for (int g = 0; g < 5; ++g) ga[g] = edge_attr[e * 5 + g];
  float ea[64];
#pragma unroll
  for (int j = 0; j < 64; ++j) {
    float a = short_b[j];
#pragma unroll
    for (int g = 0; g < 5; ++g) a += ga[g] * short_w[g * 64 + j];
    ea[j] = fmaxf(a, 0.0f);
  }
#pragma unroll
  for (int k = 0; k < 16; ++k) {
    float a = nn1_b[k];
#pragma unroll
    for (int j = 0; j < 64; ++j) a += ea[j] * nn1_w[j * 16 + k];
    tfeat[e * 16 + k] = fmaxf(a, 0.0f);
  }
}

// --------------------------- out0 = relu(h@lin0+b) -------------------------

__global__ void __launch_bounds__(32)
lin0_kernel(const _Float16* h16, const _Float16* lin0_cm, const float* lin0_b,
            float* outF, _Float16* out16, float* hidF, _Float16* hid16) {
  int lane = threadIdx.x;
  int n0 = blockIdx.x * 16;
  int nl = lane & 15, mhi = (lane >> 4) & 1;
  v16h a0 = frag_a_rows(h16, NF, n0, 0, lane);
  v16h a1 = frag_a_rows(h16, NF, n0, 32, lane);
#pragma unroll
  for (int gs = 0; gs < 4; ++gs) {
    v8f c = {};
    c = wmma_f16(a0, frag_b_cm(lin0_cm, gs * 16, 0, lane), c);
    c = wmma_f16(a1, frag_b_cm(lin0_cm, gs * 16, 32, lane), c);
    int g = gs * 16 + nl;
    float b = lin0_b[g];
#pragma unroll
    for (int v = 0; v < 8; ++v) {
      int row = n0 + v + 8 * mhi;
      float val = fmaxf(c[v] + b, 0.0f);
      _Float16 hv = (_Float16)val;
      outF[row * NF + g] = val;  hidF[row * NF + g] = val;
      out16[row * NF + g] = hv;  hid16[row * NF + g] = hv;
    }
  }
}

// ------------------- edge messages + scatter-add into agg ------------------

__global__ void __launch_bounds__(32)
edge_msg(const _Float16* x16, const int* src, const int* dst,
         const float* tfeat, const _Float16* W2p_cm, const _Float16* B2_cm,
         float* agg, int nE) {
  __shared__ int sS[32], sD[32];
  __shared__ float sT[32 * 16];
  int lane = threadIdx.x;
  int tile0 = blockIdx.x * ETILES;
  int eg = tile0 * 16 + lane;
  if (eg < nE) {
    sS[lane] = src[eg];
    sD[lane] = dst[eg];
#pragma unroll
    for (int k = 0; k < 16; ++k) sT[lane * 16 + k] = tfeat[eg * 16 + k];
  } else {
    sS[lane] = 0; sD[lane] = 0;
#pragma unroll
    for (int k = 0; k < 16; ++k) sT[lane * 16 + k] = 0.0f;
  }
  __syncthreads();
  int totTiles = (nE + 15) >> 4;
  int nt = totTiles - tile0; if (nt > ETILES) nt = ETILES;   // wave-uniform

  int nl = lane & 15, mhi = (lane >> 4) & 1;

  v16h a0[ETILES], a1[ETILES];
#pragma unroll
  for (int tt = 0; tt < ETILES; ++tt) if (tt < nt) {
    a0[tt] = frag_a_gather(x16, sS + tt * 16, 0, lane);
    a1[tt] = frag_a_gather(x16, sS + tt * 16, 32, lane);
  }

  // bias term: msg = x_j @ B2
  v8f macc[ETILES][4];
#pragma unroll
  for (int gs = 0; gs < 4; ++gs) {
    v16h bb0 = frag_b_cm(B2_cm, gs * 16, 0, lane);
    v16h bb1 = frag_b_cm(B2_cm, gs * 16, 32, lane);
#pragma unroll
    for (int tt = 0; tt < ETILES; ++tt) if (tt < nt) {
      v8f c = {};
      c = wmma_f16(a0[tt], bb0, c);
      c = wmma_f16(a1[tt], bb1, c);
      macc[tt][gs] = c;
    }
  }

  // main term: Y_k = x_j @ W2_k ; msg += t[e,k] * Y_k
#pragma unroll 1
  for (int k = 0; k < 16; ++k) {
#pragma unroll
    for (int gs = 0; gs < 4; ++gs) {
      int c0 = k * 64 + gs * 16;
      v16h b0 = frag_b_cm(W2p_cm, c0, 0, lane);
      v16h b1 = frag_b_cm(W2p_cm, c0, 32, lane);
#pragma unroll
      for (int tt = 0; tt < ETILES; ++tt) if (tt < nt) {
        v8f y = {};
        y = wmma_f16(a0[tt], b0, y);
        y = wmma_f16(a1[tt], b1, y);
#pragma unroll
        for (int v = 0; v < 8; ++v) {
          int m = v + 8 * mhi;
          macc[tt][gs][v] += sT[(tt * 16 + m) * 16 + k] * y[v];
        }
      }
    }
  }

  // scatter-add (segment_sum over dst)
#pragma unroll
  for (int tt = 0; tt < ETILES; ++tt) if (tt < nt) {
#pragma unroll
    for (int v = 0; v < 8; ++v) {
      int m = v + 8 * mhi;
      int d = sD[tt * 16 + m];
#pragma unroll
      for (int gs = 0; gs < 4; ++gs)
        atomicAdd(&agg[(size_t)d * NF + gs * 16 + nl], macc[tt][gs][v]);
    }
  }
}

// ----- node update: conv = agg/cnt + out@root + b; m = relu; GRU(m, h) -----

__global__ void __launch_bounds__(32)
node_update(const _Float16* x16_in, const float* agg, const float* cnt,
            const float* conv_bias, const _Float16* root_cm,
            const _Float16* wih_cm, const _Float16* whh_cm,
            const float* b_ih, const float* b_hh,
            float* hidF, _Float16* hid16,
            float* outF_dst, _Float16* out16_dst) {
  __shared__ __align__(16) _Float16 m16[16 * NF];
  int lane = threadIdx.x;
  int n0 = blockIdx.x * 16;
  int nl = lane & 15, mhi = (lane >> 4) & 1;

  v16h ax0 = frag_a_rows(x16_in, NF, n0, 0, lane);
  v16h ax1 = frag_a_rows(x16_in, NF, n0, 32, lane);

  // conv + relu -> m (staged in LDS to re-enter A-fragment layout)
#pragma unroll
  for (int gs = 0; gs < 4; ++gs) {
    v8f c = {};
    c = wmma_f16(ax0, frag_b_cm(root_cm, gs * 16, 0, lane), c);
    c = wmma_f16(ax1, frag_b_cm(root_cm, gs * 16, 32, lane), c);
    int g = gs * 16 + nl;
    float b = conv_bias[g];
#pragma unroll
    for (int v = 0; v < 8; ++v) {
      int m = v + 8 * mhi, row = n0 + m;
      float cn = fmaxf(cnt[row], 1.0f);
      float val = fmaxf(c[v] + agg[row * NF + g] / cn + b, 0.0f);
      m16[m * NF + g] = (_Float16)val;
    }
  }
  __syncthreads();

  v16h am0 = frag_a_rows(m16, NF, 0, 0, lane);
  v16h am1 = frag_a_rows(m16, NF, 0, 32, lane);
  v16h ah0 = frag_a_rows(hid16, NF, n0, 0, lane);
  v16h ah1 = frag_a_rows(hid16, NF, n0, 32, lane);

  v8f racc[4], zacc[4];
  // r gates: gate columns 0..63
#pragma unroll
  for (int ct = 0; ct < 4; ++ct) {
    v8f gi = {}, gh = {};
    gi = wmma_f16(am0, frag_b_cm(wih_cm, ct * 16, 0, lane), gi);
    gi = wmma_f16(am1, frag_b_cm(wih_cm, ct * 16, 32, lane), gi);
    gh = wmma_f16(ah0, frag_b_cm(whh_cm, ct * 16, 0, lane), gh);
    gh = wmma_f16(ah1, frag_b_cm(whh_cm, ct * 16, 32, lane), gh);
    int g = ct * 16 + nl;
    float bi = b_ih[g], bh = b_hh[g];
#pragma unroll
    for (int v = 0; v < 8; ++v)
      racc[ct][v] = 1.0f / (1.0f + expf(-(gi[v] + bi + gh[v] + bh)));
  }
  // z gates: gate columns 64..127
#pragma unroll
  for (int ct = 0; ct < 4; ++ct) {
    int c0 = 64 + ct * 16;
    v8f gi = {}, gh = {};
    gi = wmma_f16(am0, frag_b_cm(wih_cm, c0, 0, lane), gi);
    gi = wmma_f16(am1, frag_b_cm(wih_cm, c0, 32, lane), gi);
    gh = wmma_f16(ah0, frag_b_cm(whh_cm, c0, 0, lane), gh);
    gh = wmma_f16(ah1, frag_b_cm(whh_cm, c0, 32, lane), gh);
    int g = c0 + nl;
    float bi = b_ih[g], bh = b_hh[g];
#pragma unroll
    for (int v = 0; v < 8; ++v)
      zacc[ct][v] = 1.0f / (1.0f + expf(-(gi[v] + bi + gh[v] + bh)));
  }
  // n gates + combine: gate columns 128..191 -> output columns 0..63
#pragma unroll
  for (int ct = 0; ct < 4; ++ct) {
    int c0 = 128 + ct * 16;
    v8f gi = {}, gh = {};
    gi = wmma_f16(am0, frag_b_cm(wih_cm, c0, 0, lane), gi);
    gi = wmma_f16(am1, frag_b_cm(wih_cm, c0, 32, lane), gi);
    gh = wmma_f16(ah0, frag_b_cm(whh_cm, c0, 0, lane), gh);
    gh = wmma_f16(ah1, frag_b_cm(whh_cm, c0, 32, lane), gh);
    int g = c0 + nl;
    float bi = b_ih[g], bh = b_hh[g];
    int j = ct * 16 + nl;
#pragma unroll
    for (int v = 0; v < 8; ++v) {
      int m = v + 8 * mhi, row = n0 + m;
      float n = tanhf(gi[v] + bi + racc[ct][v] * (gh[v] + bh));
      float z = zacc[ct][v];
      float hold = hidF[row * NF + j];
      float hnew = (1.0f - z) * n + z * hold;
      _Float16 hv = (_Float16)hnew;
      hidF[row * NF + j] = hnew;   hid16[row * NF + j] = hv;
      outF_dst[row * NF + j] = hnew;  out16_dst[row * NF + j] = hv;
    }
  }
}

// ------------------------------- launcher ----------------------------------

extern "C" void kernel_launch(void* const* d_in, const int* in_sizes, int n_in,
                              void* d_out, int out_size, void* d_ws,
                              size_t ws_size, hipStream_t stream) {
  (void)n_in; (void)out_size; (void)ws_size;
  const float* h         = (const float*)d_in[0];
  const int*   ei        = (const int*)d_in[1];
  /* d_in[2] = edge_weight: unused by the reference */
  const float* edge_attr = (const float*)d_in[3];
  const float* lin0_w    = (const float*)d_in[4];
  const float* lin0_b    = (const float*)d_in[5];
  const float* short_w   = (const float*)d_in[6];
  const float* short_b   = (const float*)d_in[7];
  const float* nn1_w     = (const float*)d_in[8];
  const float* nn1_b     = (const float*)d_in[9];
  const float* nn2_w     = (const float*)d_in[10];
  const float* nn2_b     = (const float*)d_in[11];
  const float* root_w    = (const float*)d_in[12];
  const float* conv_b    = (const float*)d_in[13];
  const float* w_ih      = (const float*)d_in[14];
  const float* w_hh      = (const float*)d_in[15];
  const float* b_ih      = (const float*)d_in[16];
  const float* b_hh      = (const float*)d_in[17];

  const int N = in_sizes[0] / NF;   // 20000 nodes
  const int E = in_sizes[1] / 2;    // 50000 edges
  const int* src = ei;
  const int* dst = ei + E;

  // carve workspace (all offsets 256B-aligned)
  char* ws = (char*)d_ws;
  size_t off = 0;
  auto carve = [&](size_t bytes) -> void* {
    void* p = ws + off;
    off += (bytes + 255) & ~(size_t)255;
    return p;
  };
  _Float16* h16    = (_Float16*)carve((size_t)N * NF * 2);
  float*    outF   = (float*)   carve((size_t)N * NF * 4);
  _Float16* out16  = (_Float16*)carve((size_t)N * NF * 2);
  float*    hidF   = (float*)   carve((size_t)N * NF * 4);
  _Float16* hid16  = (_Float16*)carve((size_t)N * NF * 2);
  float*    agg    = (float*)   carve((size_t)N * NF * 4);
  float*    cnt    = (float*)   carve((size_t)N * 4);
  float*    tfeat  = (float*)   carve((size_t)E * 16 * 4);
  _Float16* W2pcm  = (_Float16*)carve((size_t)1024 * 64 * 2);
  _Float16* B2cm   = (_Float16*)carve((size_t)64 * 64 * 2);
  _Float16* lin0cm = (_Float16*)carve((size_t)64 * 64 * 2);
  _Float16* rootcm = (_Float16*)carve((size_t)64 * 64 * 2);
  _Float16* wihcm  = (_Float16*)carve((size_t)192 * 64 * 2);
  _Float16* whhcm  = (_Float16*)carve((size_t)192 * 64 * 2);

  prep_weights<<<64, 256, 0, stream>>>(lin0_w, root_w, nn2_w, nn2_b, w_ih, w_hh,
                                       lin0cm, rootcm, W2pcm, B2cm, wihcm, whhcm);
  cvt_f16<<<(N * NF + 255) / 256, 256, 0, stream>>>(h, h16, N * NF);
  zero_f32<<<(N + 255) / 256, 256, 0, stream>>>(cnt, N);
  count_dst<<<(E + 255) / 256, 256, 0, stream>>>(dst, cnt, E);
  edge_feat<<<(E + 127) / 128, 128, 0, stream>>>(edge_attr, short_w, short_b,
                                                 nn1_w, nn1_b, tfeat, E);
  lin0_kernel<<<N / 16, 32, 0, stream>>>(h16, lin0cm, lin0_b,
                                         outF, out16, hidF, hid16);

  int totTiles = (E + 15) / 16;
  int egrid = (totTiles + ETILES - 1) / ETILES;
  for (int it = 0; it < 2; ++it) {
    zero_f32<<<(N * NF + 255) / 256, 256, 0, stream>>>(agg, N * NF);
    edge_msg<<<egrid, 32, 0, stream>>>(out16, src, dst, tfeat, W2pcm, B2cm,
                                       agg, E);
    float* odst = (it == 1) ? (float*)d_out : outF;
    node_update<<<N / 16, 32, 0, stream>>>(out16, agg, cnt, conv_b, rootcm,
                                           wihcm, whhcm, b_ih, b_hh,
                                           hidF, hid16, odst, out16);
  }
}
